// GraphAttentionLayer_19189913878650
// MI455X (gfx1250) — compile-verified, matrix operand order checked by reading, not run on hardware
//
#include <hip/hip_runtime.h>
#include <hip/hip_bf16.h>

// GAT layer for MI455X (gfx1250): f16 WMMA with f32 accumulate.
// Pipeline: pack W -> GEMM1 (Wh=h@W) -> scores s1,s2 -> pack Wh -> fused
// softmax(+mask,+leaky) + GEMM2 (h' = att @ Wh) with att staged in LDS as f16.

typedef __attribute__((ext_vector_type(16))) _Float16 v16h;
typedef __attribute__((ext_vector_type(8)))  _Float16 v8h;
typedef __attribute__((ext_vector_type(8)))  float    v8f;

#define GAT_ALPHA 0.2f

static __device__ __forceinline__ v16h combine16(v8h lo, v8h hi) {
    return __builtin_shufflevector(lo, hi, 0, 1, 2, 3, 4, 5, 6, 7,
                                           8, 9, 10, 11, 12, 13, 14, 15);
}

static __device__ __forceinline__ v8f wmma_f16(v16h a, v16h b, v8f c) {
    // (neg_a, A, neg_b, B, c_mod, C, reuse_a, reuse_b)
    return __builtin_amdgcn_wmma_f32_16x16x32_f16(false, a, false, b,
                                                  (short)0, c, false, false);
}

// ---------------------------------------------------------------------------
// Pack W (256x256 f32, row-major K x N) into WMMA-B-ready f16 blocks:
// wpack[((nb*8 + kb)*32 + lane)*16 + h], h-th half holds
// K = kb*32 + (h>>3)*16 + (lane>>4)*8 + (h&7), N = nb*16 + (lane&15).
// 4096 threads total.
// ---------------------------------------------------------------------------
__global__ void gat_pack_W(const float* __restrict__ W, _Float16* __restrict__ wpack) {
    int tid   = blockIdx.x * blockDim.x + threadIdx.x;   // 0..4095
    int lane  = tid & 31;
    int kb    = (tid >> 5) & 7;
    int nb    = tid >> 8;
    int n     = nb * 16 + (lane & 15);
    int khalf = lane >> 4;
    _Float16* dst = wpack + ((size_t)((nb * 8 + kb) * 32 + lane)) * 16;
#pragma unroll
    for (int hh = 0; hh < 16; ++hh) {
        int k = kb * 32 + (hh >> 3) * 16 + khalf * 8 + (hh & 7);
        dst[hh] = (_Float16)W[(size_t)k * 256 + n];
    }
}

// ---------------------------------------------------------------------------
// GEMM1: Wh[16384][256] = h[16384][256] @ W[256][256].
// One wave per 16x16 tile; 8 waves per block; grid 2048.
// A: loaded from f32 h (two contiguous 8-float runs / lane), converted to f16.
// B: two global_load_b128 per lane from wpack.
// ---------------------------------------------------------------------------
__global__ void gat_gemm1(const float* __restrict__ h,
                          const _Float16* __restrict__ wpack,
                          float* __restrict__ wh) {
    int lane  = threadIdx.x & 31;
    int wave  = threadIdx.x >> 5;
    int tile  = blockIdx.x * 8 + wave;       // 0..16383
    int itile = tile >> 4;                   // 0..1023  (row tile of 16)
    int nb    = tile & 15;                   // 0..15    (col tile of 16)
    int khalf = lane >> 4;
    int mrow  = lane & 15;

    const float* arow = h + (size_t)(itile * 16 + mrow) * 256;
    v8f c = {};
#pragma unroll
    for (int kb = 0; kb < 8; ++kb) {
        int k0 = kb * 32 + khalf * 8;
        v8f r0 = *(const v8f*)(arow + k0);        // K = k0 .. k0+7
        v8f r1 = *(const v8f*)(arow + k0 + 16);   // K = k0+16 .. k0+23
        v16h A;
#pragma unroll
        for (int t = 0; t < 8; ++t) { A[t] = (_Float16)r0[t]; A[t + 8] = (_Float16)r1[t]; }
        const _Float16* bp = wpack + ((size_t)((nb * 8 + kb) * 32 + lane)) * 16;
        v16h B = combine16(*(const v8h*)bp, *(const v8h*)(bp + 8));
        c = wmma_f16(A, B, c);
    }
    // D layout: lane -> n = lane&15 ; VGPR v -> m = v + 8*(lane>>4)
    float* orow = wh + (size_t)itile * 16 * 256 + (size_t)nb * 16 + (lane & 15);
#pragma unroll
    for (int v = 0; v < 8; ++v)
        orow[(size_t)(v + 8 * khalf) * 256] = c[v];
}

// ---------------------------------------------------------------------------
// s1[r] = Wh[r,:] . a[0:256] ; s2[r] = Wh[r,:] . a[256:512]  (r = 0..16383)
// ---------------------------------------------------------------------------
__global__ void gat_scores(const float* __restrict__ wh, const float* __restrict__ a,
                           float* __restrict__ s1, float* __restrict__ s2) {
    int r = blockIdx.x * blockDim.x + threadIdx.x;
    const float* row = wh + (size_t)r * 256;
    float acc1 = 0.f, acc2 = 0.f;
#pragma unroll 4
    for (int k = 0; k < 256; ++k) {
        float v = row[k];
        acc1 += v * a[k];
        acc2 += v * a[256 + k];
    }
    s1[r] = acc1;
    s2[r] = acc2;
}

// ---------------------------------------------------------------------------
// Pack Wh (per batch, 2048 x 256) into WMMA-B-ready f16 blocks for GEMM2:
// whpack[(((b*16+nb)*64+kb)*32+lane)*16 + h], half h holds
// K(=j) = kb*32 + (h>>3)*16 + (lane>>4)*8 + (h&7), N(=o) = nb*16 + (lane&15).
// 262144 threads total.
// ---------------------------------------------------------------------------
__global__ void gat_pack_Wh(const float* __restrict__ wh, _Float16* __restrict__ whpack) {
    int tid   = blockIdx.x * blockDim.x + threadIdx.x;   // 0..262143
    int lane  = tid & 31;
    int kb    = (tid >> 5) & 63;
    int nb    = (tid >> 11) & 15;
    int b     = tid >> 15;
    int n     = nb * 16 + (lane & 15);
    int khalf = lane >> 4;
    _Float16* dst = whpack + ((size_t)(((b * 16 + nb) * 64 + kb) * 32 + lane)) * 16;
#pragma unroll
    for (int hh = 0; hh < 16; ++hh) {
        int j = kb * 32 + (hh >> 3) * 16 + khalf * 8 + (hh & 7);
        dst[hh] = (_Float16)wh[((size_t)b * 2048 + j) * 256 + n];
    }
}

// ---------------------------------------------------------------------------
// Fused attention + GEMM2. One block = (batch b, 16-row tile i0..i0+15).
// Phase A: flash-style softmax (online max/sum, then rebuild) of
//          e_ij = leaky(s1_i + s2_j) masked by adj; probabilities -> LDS f16.
// Phase B: h'[i0..i0+15, :] = att(16x2048) @ Wh(2048x256) via WMMA,
//          A from LDS (padded rows -> conflict-free ds_load_b128),
//          B from whpack (contiguous global_load_b128), scaled by 1/rowsum.
// ---------------------------------------------------------------------------
__global__ void gat_attention(const float* __restrict__ adj,
                              const float* __restrict__ s1g,
                              const float* __restrict__ s2g,
                              const _Float16* __restrict__ whpack,
                              float* __restrict__ out) {
    extern __shared__ char smem[];
    const int RS = 2056;                         // padded row stride (halves)
    _Float16* att  = (_Float16*)smem;            // [16][RS]
    float*    rinv = (float*)(smem + 16 * RS * 2);

    int lane = threadIdx.x & 31;
    int wave = threadIdx.x >> 5;                 // 0..7
    int i0   = blockIdx.x * 16;
    int b    = blockIdx.y;
    const float* s2b = s2g + (size_t)b * 2048;

    // ---- Phase A: each wave handles 2 rows ----
    for (int rr = 0; rr < 2; ++rr) {
        int ii = wave * 2 + rr;
        int i  = i0 + ii;
        float s1i = s1g[(size_t)b * 2048 + i];
        const float* arow = adj + (size_t)i * 2048;

        float m = -1e30f, s = 0.f;
        for (int jt = 0; jt < 2048; jt += 32) {
            int j = jt + lane;
            float e = s1i + s2b[j];
            e = e >= 0.f ? e : GAT_ALPHA * e;
            if (arow[j] != 0.f) {
                float nm = fmaxf(m, e);
                s = s * __expf(m - nm) + __expf(e - nm);
                m = nm;
            }
        }
#pragma unroll
        for (int off = 16; off > 0; off >>= 1) {
            float mo = __shfl_xor(m, off, 32);
            float so = __shfl_xor(s, off, 32);
            float nm = fmaxf(m, mo);
            s = s * __expf(m - nm) + so * __expf(mo - nm);
            m = nm;
        }
        for (int jt = 0; jt < 2048; jt += 32) {
            int j = jt + lane;
            float e = s1i + s2b[j];
            e = e >= 0.f ? e : GAT_ALPHA * e;
            float p = (arow[j] != 0.f) ? __expf(e - m) : 0.f;
            att[ii * RS + j] = (_Float16)p;
        }
        if (lane == 0) rinv[ii] = 1.f / s;
    }
    __syncthreads();

    // ---- Phase B: each wave computes 2 col tiles (16 cols each) ----
    int khalf = lane >> 4;
    int mr    = lane & 15;
    int nb0   = wave * 2;
    v8f c0 = {}, c1 = {};
    const _Float16* bbase = whpack + (size_t)b * 16 * 64 * 32 * 16;
    for (int kb = 0; kb < 64; ++kb) {
        const _Float16* ap = att + mr * RS + kb * 32 + khalf * 8;
        v16h A = combine16(*(const v8h*)ap, *(const v8h*)(ap + 16));
        const _Float16* bp0 = bbase + ((size_t)(nb0 * 64 + kb) * 32 + lane) * 16;
        const _Float16* bp1 = bp0 + (size_t)64 * 32 * 16;
        v16h B0 = combine16(*(const v8h*)bp0, *(const v8h*)(bp0 + 8));
        v16h B1 = combine16(*(const v8h*)bp1, *(const v8h*)(bp1 + 8));
        c0 = wmma_f16(A, B0, c0);
        c1 = wmma_f16(A, B1, c1);
    }
    float* ob = out + ((size_t)b * 2048 + i0) * 256;
    int n = lane & 15;
#pragma unroll
    for (int v = 0; v < 8; ++v) {
        int mrow = v + 8 * khalf;
        float sc = rinv[mrow];
        ob[(size_t)mrow * 256 + (size_t)nb0 * 16 + n]       = c0[v] * sc;
        ob[(size_t)mrow * 256 + (size_t)(nb0 + 1) * 16 + n] = c1[v] * sc;
    }
}

// ---------------------------------------------------------------------------
// Workspace layout (bytes):
//   [0,        16 MB)  Wh  f32  (16384 x 256)
//   [16 MB,    24 MB)  WhPack f16 (8 x 16 x 64 x 32 x 16 halves)
//   [24 MB,  +128 KB)  WPack  f16 (16 x 8 x 32 x 16 halves)
//   then s1 (64 KB), s2 (64 KB).   Total ~24.3 MB.
// ---------------------------------------------------------------------------
extern "C" void kernel_launch(void* const* d_in, const int* in_sizes, int n_in,
                              void* d_out, int out_size, void* d_ws, size_t ws_size,
                              hipStream_t stream) {
    (void)in_sizes; (void)n_in; (void)out_size; (void)ws_size;
    const float* h   = (const float*)d_in[0];   // (8, 2048, 256)
    const float* adj = (const float*)d_in[1];   // (2048, 2048)
    const float* W   = (const float*)d_in[2];   // (256, 256)
    const float* a   = (const float*)d_in[3];   // (512, 1)
    float* out = (float*)d_out;                 // (8, 2048, 256)

    char* ws = (char*)d_ws;
    float*    wh     = (float*)(ws);
    _Float16* whpack = (_Float16*)(ws + (size_t)16 * 1024 * 1024);
    _Float16* wpack  = (_Float16*)(ws + (size_t)24 * 1024 * 1024);
    float*    s1     = (float*)(ws + (size_t)24 * 1024 * 1024 + 131072);
    float*    s2     = (float*)(ws + (size_t)24 * 1024 * 1024 + 131072 + 65536);

    gat_pack_W <<<16,   256, 0, stream>>>(W, wpack);
    gat_gemm1  <<<2048, 256, 0, stream>>>(h, wpack, wh);
    gat_scores <<<64,   256, 0, stream>>>(wh, a, s1, s2);
    gat_pack_Wh<<<1024, 256, 0, stream>>>(wh, whpack);

    dim3 grid(128, 8);
    size_t lds = (size_t)16 * 2056 * 2 + 64;    // att f16 + row scales
    gat_attention<<<grid, 256, lds, stream>>>(adj, s1, s2, whpack, out);
}
